// BiLSTMCRF_24618752540675
// MI455X (gfx1250) — matrix operand order repaired and checked by reading, not compiled, not run on hardware
//
#include <hip/hip_runtime.h>
#include <hip/hip_bf16.h>

// ---------------------------------------------------------------------------
// BiLSTM-CRF for MI455X (gfx1250, wave32, WMMA)
//   V=50000 E=512 H=256 L=8 LAYERS=2 B=64 T=1024
// ---------------------------------------------------------------------------

#define Bsz   64
#define Tsz   1024
#define Esz   512
#define Hsz   256
#define G4    1024          // 4*H
#define KDIM  512           // E == 2*H == 512 for both layers
#define Ltag  8
#define BT    (Bsz * Tsz)   // 65536

typedef __attribute__((ext_vector_type(16))) __bf16 v16bf;
typedef __attribute__((ext_vector_type(8)))  float  v8f;

union Frag32 {            // 32-byte WMMA operand fragment (16 bf16 per lane)
    uint4  q[2];
    v16bf  v;
};

// CDNA5 async global->LDS path (ASYNCcnt-tracked).
// Builtin signature (from hipcc diagnostic): first param is
//   int __attribute__((vector_size(16))) __device__ *   i.e. v4i addrspace(1)*
#if defined(__gfx1250__) && defined(__has_builtin)
#  if __has_builtin(__builtin_amdgcn_global_load_async_to_lds_b128)
#    define USE_ASYNC_LDS 1
#  endif
#endif
#ifndef USE_ASYNC_LDS
#  define USE_ASYNC_LDS 0
#endif

typedef int v4i_gcc __attribute__((vector_size(16)));
typedef __attribute__((address_space(1))) v4i_gcc as1_v4i;
typedef __attribute__((address_space(3))) v4i_gcc as3_v4i;

__device__ __forceinline__ void async_wait0() {
#if defined(__gfx1250__)
#  if defined(__has_builtin) && __has_builtin(__builtin_amdgcn_s_wait_asynccnt)
    __builtin_amdgcn_s_wait_asynccnt(0);
#  else
    asm volatile("s_wait_asynccnt 0x0" ::: "memory");
#  endif
#endif
}

__device__ __forceinline__ float sigm_(float x) { return 1.0f / (1.0f + __expf(-x)); }
__device__ __forceinline__ float tanh_(float x) {
    float e = __expf(2.0f * x);
    return (e - 1.0f) / (e + 1.0f);
}

// ---------------------------------------------------------------------------
// f32 -> bf16 cast
// ---------------------------------------------------------------------------
__global__ void cast_f32_bf16(const float* __restrict__ in, __bf16* __restrict__ out, int n) {
    int i = blockIdx.x * 256 + threadIdx.x;
    if (i < n) out[i] = (__bf16)in[i];
}

// ---------------------------------------------------------------------------
// Embedding gather + cast:  x0[b,t,e] = bf16(emb[tokens[b,t], e])
// ---------------------------------------------------------------------------
__global__ void embed_gather(const float* __restrict__ emb, const int* __restrict__ tokens,
                             __bf16* __restrict__ x, int n) {
    int i = blockIdx.x * 256 + threadIdx.x;
    if (i >= n) return;
    int e  = i & (Esz - 1);
    int bt = i >> 9;
    x[i] = (__bf16)emb[(size_t)tokens[bt] * Esz + e];
}

// ---------------------------------------------------------------------------
// Input-side GEMM:  XS[bt, n] = sum_k X[bt,k] * W[n,k] + bias[n]
//   X:  [BT,  512] bf16 row-major     W: [1024, 512] bf16 row-major
//   XS: [BT, 1024] f32
// Block 256 thr (8 waves), 128x128 tile, K staged 32 at a time in LDS.
// Staging uses GLOBAL_LOAD_ASYNC_TO_LDS_B128 (VGPR-bypassing) when available.
// ---------------------------------------------------------------------------
__global__ __launch_bounds__(256)
void gemm_xw(const __bf16* __restrict__ X, const __bf16* __restrict__ W,
             const float* __restrict__ bias, float* __restrict__ XS) {
    __shared__ __attribute__((aligned(16))) __bf16 As[128 * 32];
    __shared__ __attribute__((aligned(16))) __bf16 Bs[128 * 32];

    const int tid  = threadIdx.x;
    const int lane = tid & 31;
    const int wv   = tid >> 5;
    const int wm   = wv & 1;          // 0..1 : which 64 rows
    const int wn   = wv >> 1;         // 0..3 : which 32 cols
    const int lrow = lane & 15;
    const int hi   = lane >> 4;       // half-wave id

    const size_t rowA0 = (size_t)blockIdx.x * 128;
    const size_t rowB0 = (size_t)blockIdx.y * 128;

    v8f acc[4][2];
    {
        v8f z = {};
        #pragma unroll
        for (int mt = 0; mt < 4; ++mt)
            #pragma unroll
            for (int nt = 0; nt < 2; ++nt) acc[mt][nt] = z;
    }

    const int srow = tid >> 1;            // 0..127 staging row
    const int scol = (tid & 1) * 16;      // 0 / 16 halves

    for (int kt = 0; kt < KDIM / 32; ++kt) {
        const __bf16* gA = X + (rowA0 + srow) * KDIM + kt * 32 + scol;
        const __bf16* gB = W + (rowB0 + srow) * KDIM + kt * 32 + scol;
        __bf16* lA = As + srow * 32 + scol;
        __bf16* lB = Bs + srow * 32 + scol;
#if USE_ASYNC_LDS
        __syncthreads();                       // previous iter's frag reads done
        // 32B per thread for each of A and B, direct to LDS (no VGPR bounce)
        __builtin_amdgcn_global_load_async_to_lds_b128((as1_v4i*)gA, (as3_v4i*)lA, 0, 0);
        __builtin_amdgcn_global_load_async_to_lds_b128((as1_v4i*)gA, (as3_v4i*)lA, 16, 0);
        __builtin_amdgcn_global_load_async_to_lds_b128((as1_v4i*)gB, (as3_v4i*)lB, 0, 0);
        __builtin_amdgcn_global_load_async_to_lds_b128((as1_v4i*)gB, (as3_v4i*)lB, 16, 0);
        async_wait0();                         // ASYNCcnt -> 0 for this wave
        __syncthreads();                       // tiles visible block-wide
#else
        const uint4* ga = (const uint4*)gA;
        uint4 a0 = ga[0], a1 = ga[1];
        const uint4* gb = (const uint4*)gB;
        uint4 b0 = gb[0], b1 = gb[1];
        __syncthreads();
        ((uint4*)lA)[0] = a0;
        ((uint4*)lA)[1] = a1;
        ((uint4*)lB)[0] = b0;
        ((uint4*)lB)[1] = b1;
        __syncthreads();
#endif

        // A fragments: lane<16 holds K 0..7 & 16..23, lane>=16 holds 8..15 & 24..31
        Frag32 afr[4];
        #pragma unroll
        for (int mt = 0; mt < 4; ++mt) {
            int m  = wm * 64 + mt * 16 + lrow;
            int k0 = hi ? 8 : 0;
            afr[mt].q[0] = *(const uint4*)(As + m * 32 + k0);
            afr[mt].q[1] = *(const uint4*)(As + m * 32 + k0 + 16);
        }
        // B fragments: lane<16 holds K 0..7 & 16..23, lane>=16 holds 8..15 & 24..31, N = lane%16
        Frag32 bfr[2];
        #pragma unroll
        for (int nt = 0; nt < 2; ++nt) {
            int n  = wn * 32 + nt * 16 + lrow;
            int k0 = hi ? 8 : 0;
            bfr[nt].q[0] = *(const uint4*)(Bs + n * 32 + k0);
            bfr[nt].q[1] = *(const uint4*)(Bs + n * 32 + k0 + 16);
        }
        #pragma unroll
        for (int mt = 0; mt < 4; ++mt)
            #pragma unroll
            for (int nt = 0; nt < 2; ++nt)
                acc[mt][nt] = __builtin_amdgcn_wmma_f32_16x16x32_bf16(
                    false, afr[mt].v, false, bfr[nt].v, (short)0, acc[mt][nt], false, false);
    }

    // epilogue: + bias, store f32
    #pragma unroll
    for (int nt = 0; nt < 2; ++nt) {
        int ncol = (int)rowB0 + wn * 32 + nt * 16 + lrow;
        float bv = bias[ncol];
        #pragma unroll
        for (int mt = 0; mt < 4; ++mt) {
            #pragma unroll
            for (int r = 0; r < 8; ++r) {
                int m = wm * 64 + mt * 16 + r + (hi ? 8 : 0);
                XS[(rowA0 + m) * G4 + ncol] = acc[mt][nt][r] + bv;
            }
        }
    }
}

// ---------------------------------------------------------------------------
// Recurrent scan (one direction per block; blockIdx.x = dir)
//   XS:  [BT, 1024] f32  pre-gates  (x@Wih^T + b)
//   Whh: [1024, 256] bf16 row-major (resident in 192MB L2; streamed per step)
//   h:   [64, 256] bf16 in LDS (double-synced single buffer)
//   c:   per-lane registers (same layout as WMMA D tiles)
//   Xout:[B,T,512] bf16, dir d writes columns d*256..d*256+255
// 512 threads = 16 waves; wave w owns hidden tile jt=w, all 4 gates, mt=0..3.
// ---------------------------------------------------------------------------
__global__ __launch_bounds__(512)
void lstm_scan(const float* __restrict__ XS_f, const float* __restrict__ XS_b,
               const __bf16* __restrict__ Whh_f, const __bf16* __restrict__ Whh_b,
               const int* __restrict__ lengths, __bf16* __restrict__ Xout) {
    const int dir = blockIdx.x;
    const float*  XS  = dir ? XS_b  : XS_f;
    const __bf16* Whh = dir ? Whh_b : Whh_f;

    __shared__ __attribute__((aligned(16))) __bf16 hbuf[Bsz * Hsz];   // 32 KB
    __shared__ int lenS[Bsz];

    const int tid  = threadIdx.x;
    const int lane = tid & 31;
    const int w    = tid >> 5;        // wave id == hidden tile jt (0..15)
    const int lrow = lane & 15;
    const int hi   = lane >> 4;

    for (int i = tid; i < Bsz * Hsz; i += 512) hbuf[i] = (__bf16)0.0f;
    if (tid < Bsz) { int lv = lengths[tid]; lenS[tid] = lv < 1 ? 1 : lv; }

    v8f c[4];
    {
        v8f z = {};
        #pragma unroll
        for (int mt = 0; mt < 4; ++mt) c[mt] = z;
    }
    const int j = w * 16 + lrow;      // hidden unit owned by this lane

    __syncthreads();

    for (int ts = 0; ts < Tsz; ++ts) {
        const int t = dir ? (Tsz - 1 - ts) : ts;

        v8f acc[4][4];                // [mt][gate]
        {
            v8f z = {};
            #pragma unroll
            for (int mt = 0; mt < 4; ++mt)
                #pragma unroll
                for (int g = 0; g < 4; ++g) acc[mt][g] = z;
        }

        // G = h_prev @ Whh^T  (K = 256, 8 x K32 WMMA steps)
        #pragma unroll
        for (int kt = 0; kt < Hsz / 32; ++kt) {
            Frag32 bfr[4];
            #pragma unroll
            for (int g = 0; g < 4; ++g) {
                int n  = g * Hsz + w * 16 + lrow;     // gate-major row of Whh
                int kb = kt * 32 + (hi ? 8 : 0);
                const uint4* p = (const uint4*)(Whh + (size_t)n * Hsz + kb);
                bfr[g].q[0] = p[0];
                bfr[g].q[1] = p[2];
                if (kt < Hsz / 32 - 1)                // pull next K slice into L1
                    __builtin_prefetch(Whh + (size_t)n * Hsz + kb + 32, 0, 3);
            }
            #pragma unroll
            for (int mt = 0; mt < 4; ++mt) {
                Frag32 afr;
                int m  = mt * 16 + lrow;
                int k0 = kt * 32 + (hi ? 8 : 0);
                afr.q[0] = *(const uint4*)(hbuf + m * Hsz + k0);
                afr.q[1] = *(const uint4*)(hbuf + m * Hsz + k0 + 16);
                #pragma unroll
                for (int g = 0; g < 4; ++g)
                    acc[mt][g] = __builtin_amdgcn_wmma_f32_16x16x32_bf16(
                        false, afr.v, false, bfr[g].v, (short)0, acc[mt][g], false, false);
            }
        }

        __syncthreads();   // all hbuf reads of this step complete

        // gates + state update; each lane owns fixed (b, j) cells
        #pragma unroll
        for (int mt = 0; mt < 4; ++mt) {
            #pragma unroll
            for (int r = 0; r < 8; ++r) {
                int b = mt * 16 + r + (hi ? 8 : 0);
                size_t base = ((size_t)b * Tsz + t) * G4;
                float gi = acc[mt][0][r] + XS[base + 0 * Hsz + j];
                float gf = acc[mt][1][r] + XS[base + 1 * Hsz + j];
                float gg = acc[mt][2][r] + XS[base + 2 * Hsz + j];
                float go = acc[mt][3][r] + XS[base + 3 * Hsz + j];
                float iv = sigm_(gi), fv = sigm_(gf), gv = tanh_(gg), ov = sigm_(go);
                float cn = fv * c[mt][r] + iv * gv;
                float hn = ov * tanh_(cn);
                bool  m  = t < lenS[b];
                float hp = (float)hbuf[b * Hsz + j];
                float h2 = m ? hn : hp;
                c[mt][r] = m ? cn : c[mt][r];
                hbuf[b * Hsz + j] = (__bf16)h2;
                Xout[((size_t)b * Tsz + t) * (2 * Hsz) + dir * Hsz + j] = (__bf16)h2;
            }
        }
        __syncthreads();   // h update visible before next step's reads
    }
}

// ---------------------------------------------------------------------------
// Emission projection:  Emis[bt, l] = sum_k X2[bt,k] * Wout[l,k] + bout[l]
// ---------------------------------------------------------------------------
__global__ __launch_bounds__(256)
void emis_proj(const __bf16* __restrict__ X2, const float* __restrict__ Wout,
               const float* __restrict__ bout, float* __restrict__ Emis) {
    __shared__ float Ws[Ltag * KDIM];      // 16 KB
    const int tid = threadIdx.x;
    for (int i = tid; i < Ltag * KDIM; i += 256) Ws[i] = Wout[i];
    __syncthreads();

    size_t bt = (size_t)blockIdx.x * 256 + tid;
    const __bf16* xp = X2 + bt * KDIM;

    float acc[Ltag];
    #pragma unroll
    for (int l = 0; l < Ltag; ++l) acc[l] = 0.0f;

    union H8 { uint4 q; __bf16 h[8]; };
    for (int k0 = 0; k0 < KDIM; k0 += 8) {
        H8 hx;
        hx.q = *(const uint4*)(xp + k0);
        #pragma unroll
        for (int kk = 0; kk < 8; ++kk) {
            float xv = (float)hx.h[kk];
            #pragma unroll
            for (int l = 0; l < Ltag; ++l) acc[l] += xv * Ws[l * KDIM + k0 + kk];
        }
    }
    #pragma unroll
    for (int l = 0; l < Ltag; ++l) Emis[bt * Ltag + l] = acc[l] + bout[l];
}

// ---------------------------------------------------------------------------
// CRF negative log-likelihood: one thread per batch element, block reduce.
// ---------------------------------------------------------------------------
__global__ __launch_bounds__(64)
void crf_nll(const float* __restrict__ Emis, const int* __restrict__ tags,
             const int* __restrict__ lengths, const float* __restrict__ trans,
             const float* __restrict__ start_t, const float* __restrict__ end_t,
             float* __restrict__ out) {
    __shared__ float trS[Ltag * Ltag];
    __shared__ float red[Bsz];
    const int b = threadIdx.x;
    if (b < Ltag * Ltag) trS[b] = trans[b];
    __syncthreads();

    float st[Ltag], et[Ltag];
    #pragma unroll
    for (int y = 0; y < Ltag; ++y) { st[y] = start_t[y]; et[y] = end_t[y]; }

    int lv  = lengths[b];
    int len = lv < 1 ? 1 : lv;
    const float* eb = Emis + (size_t)b * Tsz * Ltag;

    float alpha[Ltag];
    #pragma unroll
    for (int y = 0; y < Ltag; ++y) alpha[y] = eb[y] + st[y];

    int   t0    = tags[b * Tsz];
    float score = st[t0] + eb[t0];
    int   last  = t0;

    for (int t = 1; t < Tsz; ++t) {
        bool m = t < len;
        const float* e = eb + (size_t)t * Ltag;
        float na[Ltag];
        #pragma unroll
        for (int y = 0; y < Ltag; ++y) {
            float mx = -3.4e38f;
            float s[Ltag];
            #pragma unroll
            for (int x = 0; x < Ltag; ++x) {
                s[x] = alpha[x] + trS[y * Ltag + x];
                mx = fmaxf(mx, s[x]);
            }
            float sm = 0.0f;
            #pragma unroll
            for (int x = 0; x < Ltag; ++x) sm += __expf(s[x] - mx);
            na[y] = e[y] + mx + __logf(sm);
        }
        if (m) {
            #pragma unroll
            for (int y = 0; y < Ltag; ++y) alpha[y] = na[y];
        }
        int   tg = tags[b * Tsz + t];
        float s2 = score + trS[tg * Ltag + last] + e[tg];
        if (m) { score = s2; last = tg; }
    }

    float mx = -3.4e38f;
    #pragma unroll
    for (int y = 0; y < Ltag; ++y) mx = fmaxf(mx, alpha[y] + et[y]);
    float sm = 0.0f;
    #pragma unroll
    for (int y = 0; y < Ltag; ++y) sm += __expf(alpha[y] + et[y] - mx);
    float logZ = mx + __logf(sm);
    float gold = score + et[last];

    red[b] = logZ - gold;
    __syncthreads();
    if (b == 0) {
        float tot = 0.0f;
        for (int i = 0; i < Bsz; ++i) tot += red[i];
        out[0] = tot;
    }
}

// ---------------------------------------------------------------------------
// Workspace layout (bytes)
// ---------------------------------------------------------------------------
#define OFF_X0   ((size_t)0)                    // [BT,512]  bf16   64 MB
#define OFF_X1   (OFF_X0  + (size_t)BT * 512 * 2)
#define OFF_X2   (OFF_X1  + (size_t)BT * 512 * 2)
#define OFF_XS0  (OFF_X2  + (size_t)BT * 512 * 2)      // [BT,1024] f32 256 MB
#define OFF_XS1  (OFF_XS0 + (size_t)BT * 1024 * 4)
#define OFF_WIH  (OFF_XS1 + (size_t)BT * 1024 * 4)     // 2*2*1024*512 bf16
#define OFF_WHH  (OFF_WIH + (size_t)2 * 2 * 1024 * 512 * 2)
#define OFF_EMIS (OFF_WHH + (size_t)2 * 2 * 1024 * 256 * 2)

extern "C" void kernel_launch(void* const* d_in, const int* in_sizes, int n_in,
                              void* d_out, int out_size, void* d_ws, size_t ws_size,
                              hipStream_t stream) {
    const float* emb     = (const float*)d_in[0];
    const float* W_ih    = (const float*)d_in[1];
    const float* W_hh    = (const float*)d_in[2];
    const float* b_lstm  = (const float*)d_in[3];
    const float* W_out   = (const float*)d_in[4];
    const float* b_out   = (const float*)d_in[5];
    const float* trans   = (const float*)d_in[6];
    const float* start_t = (const float*)d_in[7];
    const float* end_t   = (const float*)d_in[8];
    const int*   tokens  = (const int*)d_in[9];
    const int*   tags    = (const int*)d_in[10];
    const int*   lengths = (const int*)d_in[11];
    (void)in_sizes; (void)n_in; (void)out_size; (void)ws_size;

    char* ws = (char*)d_ws;
    __bf16* x0      = (__bf16*)(ws + OFF_X0);
    __bf16* x1      = (__bf16*)(ws + OFF_X1);
    __bf16* x2      = (__bf16*)(ws + OFF_X2);
    float*  xs0     = (float*)(ws + OFF_XS0);
    float*  xs1     = (float*)(ws + OFF_XS1);
    __bf16* wih_bf  = (__bf16*)(ws + OFF_WIH);
    __bf16* whh_bf  = (__bf16*)(ws + OFF_WHH);
    float*  emis    = (float*)(ws + OFF_EMIS);
    float*  outp    = (float*)d_out;

    // 1) weight casts
    {
        int n_ih = 2 * 2 * 1024 * 512;
        cast_f32_bf16<<<(n_ih + 255) / 256, 256, 0, stream>>>(W_ih, wih_bf, n_ih);
        int n_hh = 2 * 2 * 1024 * 256;
        cast_f32_bf16<<<(n_hh + 255) / 256, 256, 0, stream>>>(W_hh, whh_bf, n_hh);
    }
    // 2) embedding gather
    {
        int n = BT * Esz;
        embed_gather<<<n / 256, 256, 0, stream>>>(emb, tokens, x0, n);
    }
    // 3) layers
    const __bf16* xin = x0;
    __bf16* xouts[2] = { x1, x2 };
    for (int l = 0; l < 2; ++l) {
        for (int d = 0; d < 2; ++d) {
            const __bf16* wp = wih_bf + (size_t)(l * 2 + d) * 1024 * 512;
            const float*  bp = b_lstm + (size_t)(l * 2 + d) * 1024;
            float* xsp = d ? xs1 : xs0;
            gemm_xw<<<dim3(BT / 128, G4 / 128), 256, 0, stream>>>(xin, wp, bp, xsp);
        }
        const __bf16* whh_f = whh_bf + (size_t)(l * 2 + 0) * 1024 * 256;
        const __bf16* whh_b = whh_bf + (size_t)(l * 2 + 1) * 1024 * 256;
        lstm_scan<<<2, 512, 0, stream>>>(xs0, xs1, whh_f, whh_b, lengths, xouts[l]);
        xin = xouts[l];
    }
    // 4) emissions
    emis_proj<<<BT / 256, 256, 0, stream>>>(x2, W_out, b_out, emis);
    // 5) CRF NLL -> scalar
    crf_nll<<<1, 64, 0, stream>>>(emis, tags, lengths, trans, start_t, end_t, outp);
}
